// VideoT5EncoderLayer_23227183136987
// MI455X (gfx1250) — matrix operand order repaired and checked by reading, not compiled, not run on hardware
//
#include <hip/hip_runtime.h>
#include <hip/hip_bf16.h>
#include <math.h>

// ---------------------------------------------------------------------------
// VideoT5EncoderLayer on MI455X (gfx1250).
// All GEMMs via v_wmma_f32_16x16x32_f16; LDS tiles stored fragment-major so
// each lane loads its whole WMMA fragment as one contiguous 32B read
// (2 x ds_load_b128) instead of 16 strided ds_load_u16 + v_mov_b16 packing.
// ---------------------------------------------------------------------------

typedef __attribute__((ext_vector_type(16))) _Float16 v16h;
typedef __attribute__((ext_vector_type(2)))  _Float16 h2;
typedef __attribute__((ext_vector_type(8)))  float    v8f;

// Fragment-major index helpers (CDNA5 ISA 7.12.2, wave32):
// A (16x32): lane = (m&15) | (((k>>3)&1)<<4), half = (k&7) | ((k&16)?8:0)
// B (32x16): lane = (n&15) | (((k>>3)&1)<<4), half = (k&7) | ((k&16)?8:0)
__device__ __forceinline__ int a_slot(int m, int k) {
  int lane = (m & 15) | (((k >> 3) & 1) << 4);
  int h = (k & 7) | ((k & 16) ? 8 : 0);
  return (lane << 4) | h;
}
__device__ __forceinline__ int b_slot(int n, int k) {
  int lane = (n & 15) | (((k >> 3) & 1) << 4);
  return (lane << 4) | ((k & 7) | ((k & 16) ? 8 : 0));
}

#define WMMA_F16(A_, B_, C_) \
  __builtin_amdgcn_wmma_f32_16x16x32_f16(false, (A_), false, (B_), (short)0, (C_), false, false)

// ---------------------------------------------------------------------------
// Generic batched WMMA GEMM: C = A * B (+bias)(+relu)(+resid)
// Block tile 128x64, 256 threads = 8 waves; wave w -> rows (w>>1)*32,
// cols (w&1)*32, four 16x16 accumulators (2 A-frags x 2 B-frags per k-step).
// Requires M%128==0, N%64==0, K%32==0 (true for every call site).
// FLAGS: 1=bias  2=relu  4=resid  8=B is N-major (transposed)
// Batch z decomposed as (z/nh, z%nh) with independent strides.
// ---------------------------------------------------------------------------
template <int FLAGS>
__global__ __launch_bounds__(256) void gemm_wmma(
    const float* __restrict__ A, const float* __restrict__ B,
    const float* __restrict__ bias, const float* __restrict__ R,
    float* __restrict__ C, int M, int N, int K, int lda, int ldb, int ldc,
    int nh, long sAb, long sAh, long sBb, long sBh, long sCb, long sCh) {
  // [rowblock(8)][lane(32)][half(16)] and [colblock(4)][lane(32)][half(16)]
  __shared__ alignas(32) _Float16 As[8 * 32 * 16];   // 8 KB
  __shared__ alignas(32) _Float16 Bs[4 * 32 * 16];   // 4 KB

  int z = blockIdx.z;
  long zb = z / nh, zh = z - zb * (long)nh;
  A += zb * sAb + zh * sAh;
  B += zb * sBb + zh * sBh;
  long coff = zb * sCb + zh * sCh;
  C += coff;
  const float* Rp = (FLAGS & 4) ? (R + coff) : C;

  int tn = blockIdx.x * 64, tm = blockIdx.y * 128;
  int tid = threadIdx.x, wave = tid >> 5, lane = tid & 31;
  int rb0 = (wave >> 1) * 2;   // A row-block pair (of 8)
  int cb0 = (wave & 1) * 2;    // B col-block pair (of 4)

  v8f acc00 = {}, acc01 = {}, acc10 = {}, acc11 = {};

  for (int k0 = 0; k0 < K; k0 += 32) {
    // ---- fill A tile: 128x32, even-k pairs -> aligned h2 stores ----
#pragma unroll
    for (int t = tid; t < 2048; t += 256) {
      int r = t >> 4;
      int c = (t & 15) * 2;
      float2 v = *(const float2*)&A[(long)(tm + r) * lda + (k0 + c)];
      h2 d = {(_Float16)v.x, (_Float16)v.y};
      *(h2*)&As[((r >> 4) * 512) + a_slot(r, c)] = d;
    }
    // ---- fill B tile: 32x64 ----
#pragma unroll
    for (int t = tid; t < 1024; t += 256) {
      int n = t >> 4;
      int k = (t & 15) * 2;
      float b0v, b1v;
      if (FLAGS & 8) {
        float2 v = *(const float2*)&B[(long)(tn + n) * ldb + (k0 + k)];
        b0v = v.x; b1v = v.y;
      } else {
        b0v = B[(long)(k0 + k) * ldb + (tn + n)];
        b1v = B[(long)(k0 + k + 1) * ldb + (tn + n)];
      }
      h2 d = {(_Float16)b0v, (_Float16)b1v};
      *(h2*)&Bs[((n >> 4) * 512) + b_slot(n, k)] = d;
    }
    // prefetch next k-tile into cache (global_prefetch_b8)
    if (k0 + 32 < K) {
      __builtin_prefetch(&A[(long)(tm + (tid >> 1)) * lda + (k0 + 32) + (tid & 1) * 16], 0, 3);
      if (FLAGS & 8)
        __builtin_prefetch(&B[(long)(tn + (tid >> 2)) * ldb + (k0 + 32) + (tid & 3) * 8], 0, 3);
      else
        __builtin_prefetch(&B[(long)(k0 + 32 + (tid >> 3)) * ldb + tn + (tid & 7) * 8], 0, 3);
    }
    __syncthreads();
    v16h a0 = *(const v16h*)&As[((rb0 + 0) * 32 + lane) << 4];
    v16h a1 = *(const v16h*)&As[((rb0 + 1) * 32 + lane) << 4];
    v16h b0 = *(const v16h*)&Bs[((cb0 + 0) * 32 + lane) << 4];
    v16h b1 = *(const v16h*)&Bs[((cb0 + 1) * 32 + lane) << 4];
    acc00 = WMMA_F16(a0, b0, acc00);
    acc01 = WMMA_F16(a0, b1, acc01);
    acc10 = WMMA_F16(a1, b0, acc10);
    acc11 = WMMA_F16(a1, b1, acc11);
    __syncthreads();
  }

  // C layout: vgpr i -> M = i + (lane&16 ? 8 : 0), N = lane&15
  int mr = tm + (wave >> 1) * 32 + ((lane & 16) ? 8 : 0);
  int n0 = tn + (wave & 1) * 32 + (lane & 15);

#define STORE_ACC(ACC_, MR_, NN_)                         \
  _Pragma("unroll") for (int i = 0; i < 8; ++i) {         \
    long row = (long)(MR_) + i;                           \
    float v = (ACC_)[i];                                  \
    if (FLAGS & 1) v += bias[(NN_)];                      \
    if (FLAGS & 2) v = fmaxf(v, 0.f);                     \
    if (FLAGS & 4) v += Rp[row * ldc + (NN_)];            \
    C[row * ldc + (NN_)] = v;                             \
  }
  STORE_ACC(acc00, mr,      n0)
  STORE_ACC(acc01, mr,      n0 + 16)
  STORE_ACC(acc10, mr + 16, n0)
  STORE_ACC(acc11, mr + 16, n0 + 16)
#undef STORE_ACC
}

// ---------------------------------------------------------------------------
// bbox bias GEMM: per (b,i):  S2[h,j] = sum_d Q[b,i,h*64+d] * bbox[i,j,b,d]
// M=12 (pad 16), K=64, N=256. One 256-thread block per (i,b); the 64x256
// bbox panel is read exactly once device-wide (bbox is the 268MB hot input).
// ---------------------------------------------------------------------------
__global__ __launch_bounds__(256) void bbox_scores(
    const float* __restrict__ q, const float* __restrict__ bbox,
    float* __restrict__ scores) {
  // [kstep(2)][lane(32)][half(16)] / [kstep(2)][colblock(16)][lane(32)][half(16)]
  __shared__ alignas(32) _Float16 As[2 * 32 * 16];         // 2 KB
  __shared__ alignas(32) _Float16 Bs[2 * 16 * 32 * 16];    // 32 KB
  int i = blockIdx.x, b = blockIdx.y;
  int tid = threadIdx.x;

  const float* qr = q + ((long)(b * 256 + i)) * 768;
#pragma unroll
  for (int t = tid; t < 512; t += 256) {
    int hh = t >> 5;            // 0..15 (head, >=12 zero pad)
    int d = (t & 31) * 2;       // 0..62
    float x0 = 0.f, x1 = 0.f;
    if (hh < 12) { float2 v = *(const float2*)&qr[hh * 64 + d]; x0 = v.x; x1 = v.y; }
    h2 dd = {(_Float16)x0, (_Float16)x1};
    *(h2*)&As[((d >> 5) * 512) + a_slot(hh, d & 31)] = dd;
  }
  // bbox[i, j, b, d] : offset i*262144 + j*1024 + b*64 + d
  const float* bp = bbox + (long)i * 262144 + (long)b * 64;
#pragma unroll
  for (int t = tid; t < 8192; t += 256) {
    int j = t >> 5;             // 0..255
    int d = (t & 31) * 2;       // 0..62
    float2 v = *(const float2*)&bp[(long)j * 1024 + d];
    h2 dd = {(_Float16)v.x, (_Float16)v.y};
    *(h2*)&Bs[(((d >> 5) * 16 + (j >> 4)) * 512) + b_slot(j, d & 31)] = dd;
  }
  __syncthreads();

  int wave = tid >> 5, lane = tid & 31;
  int cb0 = wave * 2;           // two 16-col blocks per wave -> 32 cols
  v8f acc0 = {}, acc1 = {};
#pragma unroll
  for (int ks = 0; ks < 2; ++ks) {
    v16h a  = *(const v16h*)&As[(ks * 32 + lane) << 4];
    v16h b0 = *(const v16h*)&Bs[((ks * 16 + cb0 + 0) * 32 + lane) << 4];
    v16h b1 = *(const v16h*)&Bs[((ks * 16 + cb0 + 1) * 32 + lane) << 4];
    acc0 = WMMA_F16(a, b0, acc0);
    acc1 = WMMA_F16(a, b1, acc1);
  }
  int hbase = (lane & 16) ? 8 : 0;
  int j0 = wave * 32 + (lane & 15);
#pragma unroll
  for (int t = 0; t < 8; ++t) {
    int h = hbase + t;
    if (h < 12) {
      long base = (((long)(b * 12 + h) * 256 + i)) * 256;
      scores[base + j0]      = acc0[t];
      scores[base + j0 + 16] = acc1[t];
    }
  }
}

// ---------------------------------------------------------------------------
// Depthwise 3x3 conv over (frames=8, L=256) per channel, pad 1.
// h0/h1 layout: (b=2, n=8, l=256, c=128)
// ---------------------------------------------------------------------------
__global__ __launch_bounds__(256) void conv_dw(
    const float* __restrict__ h0, const float* __restrict__ cw,
    const float* __restrict__ cb, float* __restrict__ h1) {
  int idx = blockIdx.x * 256 + threadIdx.x;   // 2^19 total
  int c = idx & 127;
  int l = (idx >> 7) & 255;
  int n = (idx >> 15) & 7;
  int b = idx >> 18;
  float s = cb[c];
#pragma unroll
  for (int dn = -1; dn <= 1; ++dn) {
#pragma unroll
    for (int dl = -1; dl <= 1; ++dl) {
      int nn = n + dn, ll = l + dl;
      if (nn >= 0 && nn < 8 && ll >= 0 && ll < 256)
        s += h0[(((long)(b * 8 + nn)) * 256 + ll) * 128 + c] *
             cw[c * 9 + (dn + 1) * 3 + (dl + 1)];
    }
  }
  h1[idx] = s;
}

// ---- LayerNorm(+residual): one wave32 per 768-row ----
__global__ __launch_bounds__(256) void ln_residual(
    const float* __restrict__ x, const float* __restrict__ w,
    const float* __restrict__ bvec, const float* __restrict__ res,
    float* __restrict__ out) {
  int row = blockIdx.x * 8 + (threadIdx.x >> 5);
  int lane = threadIdx.x & 31;
  const float* xr = x + (long)row * 768;
  float s = 0.f, s2 = 0.f;
  for (int j = lane; j < 768; j += 32) { float v = xr[j]; s += v; s2 += v * v; }
#pragma unroll
  for (int o = 16; o; o >>= 1) { s += __shfl_xor(s, o, 32); s2 += __shfl_xor(s2, o, 32); }
  float mu = s * (1.f / 768.f);
  float var = s2 * (1.f / 768.f) - mu * mu;
  float rstd = rsqrtf(var + 1e-5f);
  for (int j = lane; j < 768; j += 32)
    out[(long)row * 768 + j] =
        (xr[j] - mu) * rstd * w[j] + bvec[j] + res[(long)row * 768 + j];
}

// ---- RMSNorm: one wave32 per 768-row ----
__global__ __launch_bounds__(256) void rms_norm(
    const float* __restrict__ x, const float* __restrict__ w,
    float* __restrict__ out) {
  int row = blockIdx.x * 8 + (threadIdx.x >> 5);
  int lane = threadIdx.x & 31;
  const float* xr = x + (long)row * 768;
  float s2 = 0.f;
  for (int j = lane; j < 768; j += 32) { float v = xr[j]; s2 += v * v; }
#pragma unroll
  for (int o = 16; o; o >>= 1) s2 += __shfl_xor(s2, o, 32);
  float r = rsqrtf(s2 * (1.f / 768.f) + 1e-6f);
  for (int j = lane; j < 768; j += 32)
    out[(long)row * 768 + j] = xr[j] * r * w[j];
}

// ---- relative-position bucket bias + scale + softmax: one wave32 per row ----
__global__ __launch_bounds__(256) void softmax_bias(
    float* __restrict__ scores, const float* __restrict__ rel_bias) {
  long row = (long)blockIdx.x * 8 + (threadIdx.x >> 5);  // (b*12+h)*256 + i
  int lane = threadIdx.x & 31;
  int i = (int)(row & 255);
  int h = (int)((row >> 8) % 12);
  float* sr = scores + row * 256;
  const float inv_log16 = 1.f / logf(16.f);
  float vals[8];
  float mx = -1e30f;
#pragma unroll
  for (int t = 0; t < 8; ++t) {
    int j = lane + t * 32;
    int rel = j - i;
    int bucket = rel > 0 ? 16 : 0;
    int ar = rel < 0 ? -rel : rel;
    int add;
    if (ar < 8) add = ar;
    else {
      add = 8 + (int)(logf((float)ar * 0.125f) * inv_log16 * 8.f);
      if (add > 15) add = 15;
    }
    bucket += add;
    float v = (sr[j] + rel_bias[bucket * 12 + h]) * 0.125f;  // /sqrt(64)
    vals[t] = v;
    mx = fmaxf(mx, v);
  }
#pragma unroll
  for (int o = 16; o; o >>= 1) mx = fmaxf(mx, __shfl_xor(mx, o, 32));
  float s = 0.f;
#pragma unroll
  for (int t = 0; t < 8; ++t) { vals[t] = expf(vals[t] - mx); s += vals[t]; }
#pragma unroll
  for (int o = 16; o; o >>= 1) s += __shfl_xor(s, o, 32);
  float inv = 1.f / s;
#pragma unroll
  for (int t = 0; t < 8; ++t) sr[lane + t * 32] = vals[t] * inv;
}

// ---------------------------------------------------------------------------
extern "C" void kernel_launch(void* const* d_in, const int* in_sizes, int n_in,
                              void* d_out, int out_size, void* d_ws, size_t ws_size,
                              hipStream_t stream) {
  const float* hidden = (const float*)d_in[0];
  const float* bbox   = (const float*)d_in[1];
  const float* fc1_w  = (const float*)d_in[2];
  const float* fc1_b  = (const float*)d_in[3];
  const float* conv_w = (const float*)d_in[4];
  const float* conv_b = (const float*)d_in[5];
  const float* fc2_w  = (const float*)d_in[6];
  const float* fc2_b  = (const float*)d_in[7];
  const float* ln_ad_w= (const float*)d_in[8];
  const float* ln_ad_b= (const float*)d_in[9];
  const float* ln1_w  = (const float*)d_in[10];
  const float* q_w    = (const float*)d_in[11];
  const float* k_w    = (const float*)d_in[12];
  const float* v_w    = (const float*)d_in[13];
  const float* o_w    = (const float*)d_in[14];
  const float* rel_b  = (const float*)d_in[15];
  const float* ln2_w  = (const float*)d_in[16];
  const float* wi_w   = (const float*)d_in[17];
  const float* wo_w   = (const float*)d_in[18];
  float* out = (float*)d_out;

  // workspace layout (floats); total ~124 MB
  float* ws  = (float*)d_ws;
  const long RD = 4096L * 768;
  float* hs  = ws;                       // post-adapter / post-attn state
  float* hn  = hs + RD;                  // normed activations (also fc2 tmp)
  float* qb  = hn + RD;
  float* kb  = qb + RD;
  float* vb  = kb + RD;
  float* ctx = vb + RD;
  float* h0  = ctx + RD;                 // 4096x128
  float* h1  = h0 + 4096L * 128;         // 4096x128
  float* sc  = h1 + 4096L * 128;         // scores 192x256x256; reused as FFN hidden

  dim3 blk(256);

  // ---- TimeConvAdapter ----
  gemm_wmma<1><<<dim3(2, 32, 1), blk, 0, stream>>>(
      hidden, fc1_w, fc1_b, nullptr, h0, 4096, 128, 768, 768, 128, 128,
      1, 0, 0, 0, 0, 0, 0);
  conv_dw<<<dim3(2048), blk, 0, stream>>>(h0, conv_w, conv_b, h1);
  gemm_wmma<1><<<dim3(12, 32, 1), blk, 0, stream>>>(
      h1, fc2_w, fc2_b, nullptr, hn, 4096, 768, 128, 128, 768, 768,
      1, 0, 0, 0, 0, 0, 0);
  ln_residual<<<dim3(512), blk, 0, stream>>>(hn, ln_ad_w, ln_ad_b, hidden, hs);

  // ---- Self-attention ----
  rms_norm<<<dim3(512), blk, 0, stream>>>(hs, ln1_w, hn);
  gemm_wmma<0><<<dim3(12, 32, 1), blk, 0, stream>>>(
      hn, q_w, nullptr, nullptr, qb, 4096, 768, 768, 768, 768, 768,
      1, 0, 0, 0, 0, 0, 0);
  gemm_wmma<0><<<dim3(12, 32, 1), blk, 0, stream>>>(
      hn, k_w, nullptr, nullptr, kb, 4096, 768, 768, 768, 768, 768,
      1, 0, 0, 0, 0, 0, 0);
  gemm_wmma<0><<<dim3(12, 32, 1), blk, 0, stream>>>(
      hn, v_w, nullptr, nullptr, vb, 4096, 768, 768, 768, 768, 768,
      1, 0, 0, 0, 0, 0, 0);

  // bbox bias term into scores (reads 268MB bbox exactly once)
  bbox_scores<<<dim3(256, 16), blk, 0, stream>>>(qb, bbox, sc);

  // scores += Q K^T  (batched over b,h; B transposed; resid = bbox term)
  gemm_wmma<12><<<dim3(4, 2, 192), blk, 0, stream>>>(
      qb, kb, nullptr, sc, sc, 256, 256, 64, 768, 768, 256,
      12, 256L * 768, 64, 256L * 768, 64, 12L * 65536, 65536);

  softmax_bias<<<dim3(6144), blk, 0, stream>>>(sc, rel_b);

  // ctx = attn @ V  (batched, output written directly in (B,L,H*D) layout)
  gemm_wmma<0><<<dim3(1, 2, 192), blk, 0, stream>>>(
      sc, vb, nullptr, nullptr, ctx, 256, 64, 256, 256, 768, 768,
      12, 12L * 65536, 65536, 256L * 768, 64, 256L * 768, 64);

  // hs += ctx @ o_w
  gemm_wmma<4><<<dim3(12, 32, 1), blk, 0, stream>>>(
      ctx, o_w, nullptr, hs, hs, 4096, 768, 768, 768, 768, 768,
      1, 0, 0, 0, 0, 0, 0);

  // ---- FFN ----
  rms_norm<<<dim3(512), blk, 0, stream>>>(hs, ln2_w, hn);
  gemm_wmma<2><<<dim3(48, 32, 1), blk, 0, stream>>>(
      hn, wi_w, nullptr, nullptr, sc, 4096, 3072, 768, 768, 3072, 3072,
      1, 0, 0, 0, 0, 0, 0);
  gemm_wmma<4><<<dim3(12, 32, 1), blk, 0, stream>>>(
      sc, wo_w, nullptr, hs, out, 4096, 768, 3072, 3072, 768, 768,
      1, 0, 0, 0, 0, 0, 0);
}
